// AttentionRefinementModule_2568390443145
// MI455X (gfx1250) — matrix operand order repaired and checked by reading, not compiled, not run on hardware
//
#include <hip/hip_runtime.h>
#include <math.h>

typedef __attribute__((ext_vector_type(16))) __bf16 v16bf;
typedef __attribute__((ext_vector_type(8)))  __bf16 v8bf;
typedef __attribute__((ext_vector_type(8)))  float  v8f;

#define BB 8
#define CC 256
#define HH 64
#define WW 64
#define KTAPS 9

#define A_PITCH 264   // bf16 elems; 528B rows -> 4-dword bank stride, conflict-free b128
#define B_PITCH 40    // bf16 elems; 80B rows  -> 20-dword bank stride, conflict-free b128
#define SMEM_BYTES (64 * A_PITCH * 2 + 2 * 256 * B_PITCH * 2)   // 33792 + 40960 = 74752

// ---------------------------------------------------------------------------
// Kernel 0: transpose + convert deformable-conv weights to bf16.
//   src: w[o][c][ky][kx] f32  (256,256,3,3)
//   dst: wbf[tap][o][c]  bf16 (9,256,256)  -> contiguous in c for B-fragments
// ---------------------------------------------------------------------------
__global__ void wconv_bf16_kernel(const float* __restrict__ w,
                                  __bf16* __restrict__ wbf) {
    int i = blockIdx.x * blockDim.x + threadIdx.x;      // over 9*256*256
    if (i >= KTAPS * CC * CC) return;
    int tap = i / (CC * CC);
    int r   = i - tap * (CC * CC);
    int o   = r >> 8;
    int c   = r & 255;
    wbf[i] = (__bf16)w[(o * CC + c) * KTAPS + tap];
}

// ---------------------------------------------------------------------------
// Kernel 1: offset-predicting 3x3 conv (C=256 -> 18), pad 1.
// Block-uniform (b,o) so weight loads scalarize; x loads coalesced over w.
// ---------------------------------------------------------------------------
__global__ __launch_bounds__(256)
void offset_conv_kernel(const float* __restrict__ x,
                        const float* __restrict__ w_off,
                        const float* __restrict__ b_off,
                        float* __restrict__ off) {
    int blk  = blockIdx.x;
    int tile = blk & 15;          // 4096 / 256
    int bo   = blk >> 4;          // b*18 + o
    int o    = bo % 18;
    int b    = bo / 18;
    int pix  = tile * 256 + threadIdx.x;
    int h    = pix >> 6;
    int wcol = pix & 63;

    float acc = b_off[o];
    const float* xb = x + (size_t)b * CC * HH * WW;
    const float* wo = w_off + (size_t)o * CC * 9;

    for (int c = 0; c < CC; ++c) {
        const float* xc = xb + c * HH * WW;
        const float* wc = wo + c * 9;
#pragma unroll
        for (int ky = 0; ky < 3; ++ky) {
            int y = h + ky - 1;
            if (y < 0 || y >= HH) continue;
            const float* xr = xc + y * WW;
#pragma unroll
            for (int kx = 0; kx < 3; ++kx) {
                int xx = wcol + kx - 1;
                if (xx < 0 || xx >= WW) continue;
                acc = fmaf(xr[xx], wc[ky * 3 + kx], acc);
            }
        }
    }
    off[(size_t)bo * HH * WW + pix] = acc;
}

// ---------------------------------------------------------------------------
// Kernel 2: deformable conv as implicit GEMM on WMMA bf16.
//   One workgroup = one image row (64 pixels), 256 threads = 8 waves.
//   M-tile 64 (4 x 16), N = 256 (16 x 16), K = 9 taps x 256 ch (steps of 32).
//   Wave w: m-tile = w&3, n-tiles = (w>>2)*8 .. +7  -> 8 v8f accumulators.
//   B weight slabs are double-buffered via GLOBAL_LOAD_ASYNC_TO_LDS_B128
//   (ASYNCcnt), overlapping the L2 weight fetch with the WMMA chain.
// ---------------------------------------------------------------------------
__global__ __launch_bounds__(256)
void deform_wmma_kernel(const float* __restrict__ x,
                        const float* __restrict__ off,
                        const __bf16* __restrict__ wbf,
                        const float* __restrict__ bias,
                        float* __restrict__ out) {
    extern __shared__ __align__(16) char smem_raw[];
    __bf16 (*Alds)[A_PITCH] = (__bf16(*)[A_PITCH])smem_raw;            // 64 rows
    __bf16* Bbase = (__bf16*)(smem_raw + 64 * A_PITCH * sizeof(__bf16)); // 2 bufs

    const int blk  = blockIdx.x;       // 0..511
    const int b    = blk >> 6;
    const int h    = blk & 63;
    const int t    = threadIdx.x;
    const int lane = t & 31;
    const int wv   = t >> 5;           // wave id 0..7
    const int mi   = wv & 3;           // m-tile
    const int wc   = wv >> 2;          // n half (0/1)

    const v8f vzero = {0.f, 0.f, 0.f, 0.f, 0.f, 0.f, 0.f, 0.f};
    v8f acc[8];
#pragma unroll
    for (int j = 0; j < 8; ++j) acc[j] = vzero;

    // sampling mapping: thread -> (pixel sp, channel phase cq)
    const int sp = t & 63;
    const int cq = t >> 6;             // 0..3

    const float* xb = x + (size_t)b * CC * HH * WW;

    // per-thread constants for the async B staging (row = out-channel t)
    const unsigned voff = (unsigned)(t * CC * (int)sizeof(__bf16));  // t*512 B

    for (int tap = 0; tap < KTAPS; ++tap) {
        const int ky = tap / 3;
        const int kx = tap - 3 * ky;

        __syncthreads();   // all reads of Alds / both B buffers from prev tap done

        // ---- bilinear sampling for this tap into Alds (bf16) ----
        const float dy = off[((size_t)(b * 18 + 2 * tap) * HH + h) * WW + sp];
        const float dx = off[((size_t)(b * 18 + 2 * tap + 1) * HH + h) * WW + sp];
        const float py = (float)(h - 1 + ky) + dy;
        const float px = (float)(sp - 1 + kx) + dx;
        const float y0f = floorf(py), x0f = floorf(px);
        const int iy0 = (int)y0f, ix0 = (int)x0f;
        const int iy1 = iy0 + 1, ix1 = ix0 + 1;
        const float wy1 = py - y0f, wx1 = px - x0f;
        const float wy0 = 1.f - wy1, wx0 = 1.f - wx1;
        const bool vy0 = (iy0 >= 0) && (iy0 < HH);
        const bool vy1 = (iy1 >= 0) && (iy1 < HH);
        const bool vx0 = (ix0 >= 0) && (ix0 < WW);
        const bool vx1 = (ix1 >= 0) && (ix1 < WW);
        const float w00 = (vy0 && vx0) ? wy0 * wx0 : 0.f;
        const float w01 = (vy0 && vx1) ? wy0 * wx1 : 0.f;
        const float w10 = (vy1 && vx0) ? wy1 * wx0 : 0.f;
        const float w11 = (vy1 && vx1) ? wy1 * wx1 : 0.f;
        const int cy0 = min(max(iy0, 0), HH - 1);
        const int cy1 = min(max(iy1, 0), HH - 1);
        const int cx0 = min(max(ix0, 0), WW - 1);
        const int cx1 = min(max(ix1, 0), WW - 1);
        const int o00 = cy0 * WW + cx0;
        const int o01 = cy0 * WW + cx1;
        const int o10 = cy1 * WW + cx0;
        const int o11 = cy1 * WW + cx1;

        for (int c = cq; c < CC; c += 4) {
            const float* xc = xb + c * HH * WW;
            float v = w00 * xc[o00] + w01 * xc[o01] +
                      w10 * xc[o10] + w11 * xc[o11];
            Alds[sp][c] = (__bf16)v;
        }

        // async DMA issue of one 256x32 bf16 weight slab into B buffer `buf`.
        // GVS addressing: mem = SGPR64 + VGPR32(byte offset) + IOFFSET;
        // IOFFSET advances LDS and memory address together (ISA 15.18.3),
        // so each thread copies its contiguous 64-byte row as 4x b128.
        auto issueB = [&](int buf, int kc) {
            const unsigned long long sbase =
                (unsigned long long)(size_t)(wbf + (size_t)(tap * CC) * CC + kc * 32);
            const unsigned daddr =
                (unsigned)(size_t)(Bbase + (size_t)buf * 256 * B_PITCH +
                                   (size_t)t * B_PITCH);
            asm volatile(
                "global_load_async_to_lds_b128 %0, %1, %2 offset:0\n\t"
                "global_load_async_to_lds_b128 %0, %1, %2 offset:16\n\t"
                "global_load_async_to_lds_b128 %0, %1, %2 offset:32\n\t"
                "global_load_async_to_lds_b128 %0, %1, %2 offset:48"
                :: "v"(daddr), "v"(voff), "s"(sbase) : "memory");
        };

        issueB(0, 0);   // prologue: slice 0 in flight during nothing -> min latency

        // ---- GEMM over this tap's 256 channels in K-slices of 32 ----
        for (int kc = 0; kc < 8; ++kc) {
            const int cur = kc & 1;
            __syncthreads();   // kc==0: Alds visible; kc>0: reads of buf[cur^1] done
            if (kc < 7) {
                issueB(cur ^ 1, kc + 1);                     // prefetch next slab
                asm volatile("s_wait_asynccnt 0x4" ::: "memory");  // cur's 4 done
            } else {
                asm volatile("s_wait_asynccnt 0x0" ::: "memory");
            }
            __syncthreads();   // publish buf[cur] to all waves

            const __bf16* Bcur = Bbase + (size_t)cur * 256 * B_PITCH;

            // A fragment (16x32 bf16): lanes 0-15 -> K {0..7,16..23},
            // lanes 16-31 -> K {8..15,24..31}, per CDNA5 ISA layout.
            const int arow = mi * 16 + (lane & 15);
            const int kg   = lane >> 4;
            v8bf alo = *(const v8bf*)&Alds[arow][kc * 32 + kg * 8];
            v8bf ahi = *(const v8bf*)&Alds[arow][kc * 32 + 16 + kg * 8];
            v16bf afrag = __builtin_shufflevector(
                alo, ahi, 0, 1, 2, 3, 4, 5, 6, 7, 8, 9, 10, 11, 12, 13, 14, 15);

#pragma unroll
            for (int j = 0; j < 8; ++j) {
                // B fragment (32x16 bf16): lane = column, K packed along VGPRs
                const int n = (wc * 8 + j) * 16 + (lane & 15);
                v8bf blo = *(const v8bf*)(Bcur + (size_t)n * B_PITCH + kg * 8);
                v8bf bhi = *(const v8bf*)(Bcur + (size_t)n * B_PITCH + 16 + kg * 8);
                v16bf bfrag = __builtin_shufflevector(
                    blo, bhi, 0, 1, 2, 3, 4, 5, 6, 7, 8, 9, 10, 11, 12, 13, 14, 15);
                acc[j] = __builtin_amdgcn_wmma_f32_16x16x32_bf16(
                    false, afrag, false, bfrag, (short)0, acc[j], false, false);
            }
        }
    }

    // ---- epilogue: bias + sigmoid, straight from accumulators ----
    // C/D layout: vgpr v, lanes 0-15 -> M=v, lanes 16-31 -> M=v+8; N=lane&15
    const int pbase = mi * 16 + ((lane >> 4) << 3);
#pragma unroll
    for (int j = 0; j < 8; ++j) {
        const int o = (wc * 8 + j) * 16 + (lane & 15);
        const float bo = bias[o];
        float* op = out + ((size_t)(b * CC + o) * HH + h) * WW;
#pragma unroll
        for (int v = 0; v < 8; ++v) {
            const float val = acc[j][v] + bo;
            op[pbase + v] = 1.f / (1.f + __expf(-val));
        }
    }
}

// ---------------------------------------------------------------------------
extern "C" void kernel_launch(void* const* d_in, const int* in_sizes, int n_in,
                              void* d_out, int out_size, void* d_ws, size_t ws_size,
                              hipStream_t stream) {
    const float* x     = (const float*)d_in[0];   // [8,256,64,64]
    const float* w_off = (const float*)d_in[1];   // [18,256,3,3]
    const float* b_off = (const float*)d_in[2];   // [18]
    const float* w     = (const float*)d_in[3];   // [256,256,3,3]
    const float* bias  = (const float*)d_in[4];   // [256]
    float* out = (float*)d_out;                   // [8,256,64,64]

    // workspace: offsets (f32) then bf16-transposed weights
    float*  off_ws = (float*)d_ws;                                  // 8*18*64*64
    __bf16* wbf    = (__bf16*)((char*)d_ws +
                               (size_t)BB * 18 * HH * WW * sizeof(float));

    wconv_bf16_kernel<<<(KTAPS * CC * CC + 255) / 256, 256, 0, stream>>>(w, wbf);
    offset_conv_kernel<<<BB * 18 * 16, 256, 0, stream>>>(x, w_off, b_off, off_ws);
    deform_wmma_kernel<<<BB * HH, 256, SMEM_BYTES, stream>>>(x, off_ws, wbf, bias, out);
}
